// NonLocal2D_53377853555280
// MI455X (gfx1250) — matrix operand order repaired and checked by reading, not compiled, not run on hardware
//
#include <hip/hip_runtime.h>
#include <hip/hip_bf16.h>

// ---------------------------------------------------------------------------
// NonLocal2D (dot_product mode, no softmax) collapses algebraically:
//   out = x + U·x + beta,  U = (Wout · S^T / Nk) · Wt,  S = phi · g^T
// where phi/g are the pooled 1x1-conv features of x2.
// All heavy GEMMs run on v_wmma_f32_16x16x32_bf16 (wave32 CDNA5 WMMA).
// ---------------------------------------------------------------------------

typedef __attribute__((ext_vector_type(16))) __bf16 v16bf;
typedef __attribute__((ext_vector_type(8)))  __bf16 v8bf;
typedef __attribute__((ext_vector_type(8)))  float  v8f;

#define C_IN   256
#define C_MID  128
#define HH     96
#define NPOS   (96 * 96)   // 9216 positions per channel
#define NKP    (48 * 48)   // 2304 pooled key positions

__device__ __forceinline__ unsigned short f2bf_bits(float f) {
  unsigned u = __builtin_bit_cast(unsigned, f);
  unsigned r = u + 0x7FFFu + ((u >> 16) & 1u);   // round-to-nearest-even
  return (unsigned short)(r >> 16);
}
__device__ __forceinline__ unsigned pack2_bf(float lo, float hi) {
  return (unsigned)f2bf_bits(lo) | ((unsigned)f2bf_bits(hi) << 16);
}
__device__ __forceinline__ __bf16 f2bf(float f) {
  unsigned short s = f2bf_bits(f);
  return __builtin_bit_cast(__bf16, s);
}

// Build a 16-element bf16 fragment from two contiguous 16B chunks
// (callers use contiguous k = 16*half + s for A and B alike, not the ISA table order).
__device__ __forceinline__ v16bf load_frag_bf16(const unsigned short* p0,
                                                const unsigned short* p1) {
  v8bf lo = *reinterpret_cast<const v8bf*>(p0);
  v8bf hi = *reinterpret_cast<const v8bf*>(p1);
  v16bf r;
#pragma unroll
  for (int i = 0; i < 8; ++i) { r[i] = lo[i]; r[i + 8] = hi[i]; }
  return r;
}

// A fragment directly from f32 weights: 8 floats at +0 and 8 floats at +8.
__device__ __forceinline__ v16bf pack_w_f32(const float* w) {
  v16bf a;
#pragma unroll
  for (int i = 0; i < 8; ++i) a[i] = f2bf(w[i]);
#pragma unroll
  for (int i = 0; i < 8; ++i) a[i + 8] = f2bf(w[8 + i]);
  return a;
}

__device__ __forceinline__ v8f wmma_bf16(v16bf a, v16bf b, v8f c) {
  return __builtin_amdgcn_wmma_f32_16x16x32_bf16(
      /*neg_a=*/false, a, /*neg_b=*/false, b,
      /*c_mod=*/(short)0, c, /*reuse_a=*/false, /*reuse_b=*/false);
}

// ---------------------------------------------------------------------------
// Kernel 1: fused 1x1-conv (128x256 GEMM) + 2x2 maxpool over x2.
// One block per (pooled row, batch, weight-set). Conv computed at full res
// for two image rows (192 positions = 12 N-tiles); pooled pairs are
// (col n, n^1) across lanes -> __shfl_xor, and (tile j, j+6) across rows.
// Per wave: M-tile = wave, so A is built once per K-step and reused by all
// 12 WMMAs; B fragments are software-pipelined one j ahead.
// ---------------------------------------------------------------------------
__global__ __launch_bounds__(256) void k_conv_pool(
    const float* __restrict__ x2,
    const float* __restrict__ Wp, const float* __restrict__ bp,
    const float* __restrict__ Wg, const float* __restrict__ bg,
    unsigned short* __restrict__ phibuf, unsigned short* __restrict__ gbuf) {
  const int ph  = blockIdx.x;   // pooled row 0..47
  const int n   = blockIdx.y;   // batch
  const int sel = blockIdx.z;   // 0 -> phi (Wp), 1 -> g (Wg)
  const float* Wc = sel ? Wg : Wp;
  const float* bc = sel ? bg : bp;
  unsigned short* dst = sel ? gbuf : phibuf;

  // [pos 0..191][k-slice 0..127], stride 136 halfwords (272B, 16B aligned)
  __shared__ __align__(16) unsigned short Xs[192 * 136];

  const int tid  = threadIdx.x;
  const int lane = tid & 31;
  const int wave = tid >> 5;
  const int half = (lane >> 4) & 1;
  const int lr   = lane & 15;

  const float* xin = x2 + (size_t)n * C_IN * NPOS + (size_t)(2 * ph) * HH;

  v8f acc[6][2];
#pragma unroll
  for (int j = 0; j < 6; ++j) { acc[j][0] = 0.0f; acc[j][1] = 0.0f; }

  const int m = wave * 16 + lr;   // wave owns M-tile `wave` for all 6 units

  for (int kslice = 0; kslice < 2; ++kslice) {
    // Stage 128 channels x 192 positions of x2 as bf16 (position-major),
    // packing two channels per 32-bit LDS store.
    for (int e = tid; e < 64 * 192; e += 256) {
      const int p  = e % 192;              // coalesced over p
      const int k2 = e / 192;              // channel pair
      const size_t base =
          (size_t)(kslice * 128 + 2 * k2) * NPOS + (p / 96) * HH + (p % 96);
      const float v0 = xin[base];
      const float v1 = xin[base + NPOS];
      *reinterpret_cast<unsigned*>(&Xs[p * 136 + 2 * k2]) = pack2_bf(v0, v1);
    }
    __syncthreads();

#pragma unroll
    for (int ks = 0; ks < 4; ++ks) {
      const float* wrow =
          Wc + (size_t)m * C_IN + kslice * 128 + ks * 32 + (half ? 16 : 0);
      const v16bf a = pack_w_f32(wrow);
      const int kl = ks * 32 + (half ? 16 : 0);

      // software-pipelined B fragments (row0 tile j, row1 tile j+6)
      v16bf b0 = load_frag_bf16(&Xs[(0 * 16 + lr) * 136 + kl],
                                &Xs[(0 * 16 + lr) * 136 + kl + 8]);
      v16bf b1 = load_frag_bf16(&Xs[(6 * 16 + lr) * 136 + kl],
                                &Xs[(6 * 16 + lr) * 136 + kl + 8]);
#pragma unroll
      for (int j = 0; j < 6; ++j) {
        v16bf nb0, nb1;
        if (j < 5) {
          nb0 = load_frag_bf16(&Xs[((j + 1) * 16 + lr) * 136 + kl],
                               &Xs[((j + 1) * 16 + lr) * 136 + kl + 8]);
          nb1 = load_frag_bf16(&Xs[((j + 7) * 16 + lr) * 136 + kl],
                               &Xs[((j + 7) * 16 + lr) * 136 + kl + 8]);
        }
        acc[j][0] = wmma_bf16(a, b0, acc[j][0]);
        acc[j][1] = wmma_bf16(a, b1, acc[j][1]);
        b0 = nb0;
        b1 = nb1;
      }
    }
    __syncthreads();
  }

  // 2x2 max-pool (+bias) and store bf16 feature [c][kidx].
#pragma unroll
  for (int j = 0; j < 6; ++j) {
#pragma unroll
    for (int v = 0; v < 8; ++v) {
      const int mm = wave * 16 + v + (half ? 8 : 0);
      const float c0 = acc[j][0][v];
      const float c1 = acc[j][1][v];
      const float p0 = fmaxf(c0, __shfl_xor(c0, 1, 32));
      const float p1 = fmaxf(c1, __shfl_xor(c1, 1, 32));
      const float mx = fmaxf(p0, p1) + bc[mm];
      if ((lane & 1) == 0) {
        const int w = j * 8 + (lr >> 1);   // pooled col 0..47
        dst[((size_t)n * C_MID + mm) * NKP + ph * 48 + w] = f2bf_bits(mx);
      }
    }
  }
}

// ---------------------------------------------------------------------------
// Kernel 2: S = phi · g^T per batch  ([128x128], K = 2304). One block/batch;
// wave w owns row-tile w, loops 8 col tiles; pure global bf16 -> L2 resident.
// ---------------------------------------------------------------------------
__global__ __launch_bounds__(256) void k_gram(
    const unsigned short* __restrict__ phibuf,
    const unsigned short* __restrict__ gbuf,
    float* __restrict__ Sbuf) {
  const int n    = blockIdx.x;
  const int tid  = threadIdx.x;
  const int lane = tid & 31;
  const int wave = tid >> 5;
  const int half = (lane >> 4) & 1;
  const int lr   = lane & 15;

  const unsigned short* phv = phibuf + (size_t)n * C_MID * NKP;
  const unsigned short* ggv = gbuf   + (size_t)n * C_MID * NKP;

  v8f acc[8];
#pragma unroll
  for (int j = 0; j < 8; ++j) acc[j] = 0.0f;

  const int m = wave * 16 + lr;
  for (int ks = 0; ks < NKP / 32; ++ks) {   // 72 K-steps
    const int kA = ks * 32 + (half ? 16 : 0);
    v16bf a = load_frag_bf16(phv + (size_t)m * NKP + kA,
                             phv + (size_t)m * NKP + kA + 8);
    const int kB = ks * 32 + (half ? 16 : 0);
#pragma unroll
    for (int j = 0; j < 8; ++j) {
      const int col = j * 16 + lr;
      v16bf b = load_frag_bf16(ggv + (size_t)col * NKP + kB,
                               ggv + (size_t)col * NKP + kB + 8);
      acc[j] = wmma_bf16(a, b, acc[j]);
    }
  }

  float* S = Sbuf + (size_t)n * C_MID * C_MID;
#pragma unroll
  for (int j = 0; j < 8; ++j)
#pragma unroll
    for (int v = 0; v < 8; ++v)
      S[(size_t)(wave * 16 + v + (half ? 8 : 0)) * C_MID + j * 16 + lr] =
          acc[j][v];
}

// ---------------------------------------------------------------------------
// Kernel 3a: V[c'][i] = sum_c S[c][c'] * Wt[c][i];  vb[c'] = sum_c S[c][c']*bt[c]
// ---------------------------------------------------------------------------
__global__ __launch_bounds__(256) void k_fold1(
    const float* __restrict__ Sbuf, const float* __restrict__ Wt,
    const float* __restrict__ bt,
    float* __restrict__ Vbuf, float* __restrict__ vbbuf) {
  const int n = blockIdx.x;
  const float* S = Sbuf + (size_t)n * C_MID * C_MID;
  for (int idx = threadIdx.x; idx < C_MID * C_IN; idx += 256) {
    const int cp = idx >> 8;
    const int i  = idx & 255;
    float a = 0.f;
    for (int c = 0; c < C_MID; ++c) a += S[c * C_MID + cp] * Wt[c * C_IN + i];
    Vbuf[(size_t)n * C_MID * C_IN + idx] = a;
  }
  if (threadIdx.x < C_MID) {
    const int cp = threadIdx.x;
    float a = 0.f;
    for (int c = 0; c < C_MID; ++c) a += S[c * C_MID + cp] * bt[c];
    vbbuf[n * C_MID + cp] = a;
  }
}

// ---------------------------------------------------------------------------
// Kernel 3b: U[o][i] = (1/Nk) sum_c' Wout[o][c'] V[c'][i] (stored bf16);
//            beta[o] = (1/Nk) sum_c' Wout[o][c'] vb[c'] + bout[o]
// ---------------------------------------------------------------------------
__global__ __launch_bounds__(256) void k_fold2(
    const float* __restrict__ Vbuf, const float* __restrict__ vbbuf,
    const float* __restrict__ Wout, const float* __restrict__ bout,
    unsigned short* __restrict__ Ubf, float* __restrict__ betabuf) {
  const int n = blockIdx.x;
  const float* V  = Vbuf  + (size_t)n * C_MID * C_IN;
  const float* vb = vbbuf + n * C_MID;
  const float scale = 1.0f / (float)NKP;
  for (int idx = threadIdx.x; idx < C_IN * C_IN; idx += 256) {
    const int o = idx >> 8;
    const int i = idx & 255;
    float a = 0.f;
    for (int cp = 0; cp < C_MID; ++cp)
      a += Wout[o * C_MID + cp] * V[cp * C_IN + i];
    Ubf[(size_t)n * C_IN * C_IN + idx] = f2bf_bits(a * scale);
  }
  {
    const int o = threadIdx.x;
    float a = 0.f;
    for (int cp = 0; cp < C_MID; ++cp) a += Wout[o * C_MID + cp] * vb[cp];
    betabuf[n * C_IN + o] = a * scale + bout[o];
  }
}

// ---------------------------------------------------------------------------
// Kernel 4: out = x + U·x + beta. [256 x 256] GEMM over 9216 positions/batch.
// Block = 64-position strip; x staged to LDS as bf16 (position-major,
// packed 2-per-store). B fragments pipelined one K-step ahead; residual-x
// cachelines prefetched (global_prefetch_b8) before the WMMA loop.
// ---------------------------------------------------------------------------
__global__ __launch_bounds__(256) void k_apply(
    const float* __restrict__ x,
    const unsigned short* __restrict__ Ubf,
    const float* __restrict__ beta,
    float* __restrict__ out) {
  const int strip = blockIdx.x;   // 0..143
  const int n     = blockIdx.y;
  const int p0    = strip * 64;

  // [pos 0..63][k 0..255], stride 264 halfwords (528B, 16B aligned)
  __shared__ __align__(16) unsigned short Xs[64 * 264];

  const int tid  = threadIdx.x;
  const int lane = tid & 31;
  const int wave = tid >> 5;
  const int half = (lane >> 4) & 1;
  const int lr   = lane & 15;

  const float* xin = x + (size_t)n * C_IN * NPOS;
  const unsigned short* U = Ubf + (size_t)n * C_IN * C_IN;

  for (int e = tid; e < 64 * 128; e += 256) {
    const int pp = e & 63;                 // coalesced over positions
    const int k2 = e >> 6;                 // channel pair
    const float v0 = xin[(size_t)(2 * k2) * NPOS + p0 + pp];
    const float v1 = xin[(size_t)(2 * k2 + 1) * NPOS + p0 + pp];
    *reinterpret_cast<unsigned*>(&Xs[pp * 264 + 2 * k2]) = pack2_bf(v0, v1);
  }
  __syncthreads();

  v8f acc[8];
#pragma unroll
  for (int j = 0; j < 8; ++j) acc[j] = 0.0f;

  const int nt  = wave & 3;   // N tile (fixed per wave -> B reused 8x)
  const int mtb = wave >> 2;
  const int col = nt * 16 + lr;

  // warm the residual-x lines the epilogue will read
#pragma unroll
  for (int j = 0; j < 8; ++j)
    __builtin_prefetch(
        &xin[(size_t)((mtb + 2 * j) * 16 + lr) * NPOS + p0 + nt * 16], 0, 3);

  // software-pipelined B fragment over K-steps
  v16bf b = load_frag_bf16(&Xs[col * 264 + (half ? 16 : 0)],
                           &Xs[col * 264 + (half ? 16 : 0) + 8]);
#pragma unroll
  for (int ks = 0; ks < 8; ++ks) {
    v16bf bn;
    if (ks < 7) {
      const int kB = (ks + 1) * 32 + (half ? 16 : 0);
      bn = load_frag_bf16(&Xs[col * 264 + kB], &Xs[col * 264 + kB + 8]);
    }
    const int kA = ks * 32 + (half ? 16 : 0);
#pragma unroll
    for (int j = 0; j < 8; ++j) {
      const int m = (mtb + 2 * j) * 16 + lr;
      v16bf a = load_frag_bf16(U + (size_t)m * C_IN + kA,
                               U + (size_t)m * C_IN + kA + 8);
      acc[j] = wmma_bf16(a, b, acc[j]);
    }
    b = bn;
  }

  float* outp = out + (size_t)n * C_IN * NPOS;
#pragma unroll
  for (int j = 0; j < 8; ++j) {
    const int mt = mtb + 2 * j;
#pragma unroll
    for (int v = 0; v < 8; ++v) {
      const int m = mt * 16 + v + (half ? 8 : 0);
      const int p = p0 + nt * 16 + lr;
      outp[(size_t)m * NPOS + p] =
          xin[(size_t)m * NPOS + p] + acc[j][v] + beta[n * C_IN + m];
    }
  }
}

// ---------------------------------------------------------------------------
extern "C" void kernel_launch(void* const* d_in, const int* in_sizes, int n_in,
                              void* d_out, int out_size, void* d_ws,
                              size_t ws_size, hipStream_t stream) {
  (void)in_sizes; (void)n_in; (void)out_size; (void)ws_size;
  const float* x    = (const float*)d_in[0];
  const float* x2   = (const float*)d_in[1];
  const float* Wg   = (const float*)d_in[2];
  const float* bg   = (const float*)d_in[3];
  const float* Wt   = (const float*)d_in[4];
  const float* bt   = (const float*)d_in[5];
  const float* Wp   = (const float*)d_in[6];
  const float* bp   = (const float*)d_in[7];
  const float* Wout = (const float*)d_in[8];
  const float* bout = (const float*)d_in[9];
  float* out = (float*)d_out;

  char* wsb = (char*)d_ws;
  size_t off = 0;
  auto take = [&](size_t bytes) -> char* {
    char* p = wsb + off;
    off += (bytes + 255) & ~(size_t)255;
    return p;
  };

  unsigned short* phibuf = (unsigned short*)take((size_t)4 * C_MID * NKP * 2);
  unsigned short* gbuf   = (unsigned short*)take((size_t)4 * C_MID * NKP * 2);
  float*          Sbuf   = (float*)take((size_t)4 * C_MID * C_MID * 4);
  float*          Vbuf   = (float*)take((size_t)4 * C_MID * C_IN * 4);
  float*          vbbuf  = (float*)take((size_t)4 * C_MID * 4);
  unsigned short* Ubf    = (unsigned short*)take((size_t)4 * C_IN * C_IN * 2);
  float*          betab  = (float*)take((size_t)4 * C_IN * 4);

  k_conv_pool<<<dim3(48, 4, 2), 256, 0, stream>>>(x2, Wp, bp, Wg, bg,
                                                  phibuf, gbuf);
  k_gram<<<4, 256, 0, stream>>>(phibuf, gbuf, Sbuf);
  k_fold1<<<4, 256, 0, stream>>>(Sbuf, Wt, bt, Vbuf, vbbuf);
  k_fold2<<<4, 256, 0, stream>>>(Vbuf, vbbuf, Wout, bout, Ubf, betab);
  k_apply<<<dim3(NPOS / 64, 4), 256, 0, stream>>>(x, Ubf, betab, out);
}
